// MultiHeadAttention_10857677324519
// MI455X (gfx1250) — matrix operand order repaired and checked by reading, not compiled, not run on hardware
//
#include <hip/hip_runtime.h>
#include <hip/hip_bf16.h>

// ---------------------------------------------------------------------------
// Types for CDNA5 WMMA (wave32, V_WMMA_F32_16X16X32_BF16)
// ---------------------------------------------------------------------------
typedef __attribute__((ext_vector_type(16))) __bf16 v16bf;
typedef __attribute__((ext_vector_type(2)))  __bf16 v2bf;
typedef __attribute__((ext_vector_type(8)))  float  v8f;

union Frag16 {
    v16bf v;
    unsigned int u[8];
};

// Hardware f32 -> bf16 (RNE) conversions
__device__ __forceinline__ unsigned short f2bf(float f) {
    return __builtin_bit_cast(unsigned short, (__bf16)f);
}
__device__ __forceinline__ unsigned int pack_bf2(float a, float b) {
    v2bf p;
    p[0] = (__bf16)a;
    p[1] = (__bf16)b;
    return __builtin_bit_cast(unsigned int, p);
}

__device__ __forceinline__ v8f wmma_bf16(v16bf a, v16bf b, v8f c) {
    return __builtin_amdgcn_wmma_f32_16x16x32_bf16(
        /*neg_a=*/false, a, /*neg_b=*/false, b,
        /*c_mod=*/(short)0, c, /*reuse_a=*/false, /*reuse_b=*/false);
}

// CDNA5 async global->LDS DMA (16B per lane), tracked by ASYNCcnt.
// VDST VGPR holds the LDS byte address (low 32 bits of flat shared pointer),
// VADDR is the 64-bit global address, SADDR=off (GV mode).
__device__ __forceinline__ void async_ld_b128(void* lds, const void* gaddr) {
    unsigned int l = (unsigned int)(size_t)lds;
    asm volatile("global_load_async_to_lds_b128 %0, %1, off"
                 :: "v"(l), "v"(gaddr) : "memory");
}
// Drain all async DMAs
__device__ __forceinline__ void wait_async0() {
    asm volatile("s_wait_asynccnt 0x0" ::: "memory");
}
// Drain all but the newest 4 async DMAs (ASYNCcnt completes in order ->
// waits for the previous tile while the next tile's 4 loads stay in flight)
__device__ __forceinline__ void wait_async_le4() {
    asm volatile("s_wait_asynccnt 0x4" ::: "memory");
}

// A fragment: 16x32 (MxK) bf16 tile, row-major in LDS with stride lda.
// Lanes 0-15 -> M=lane; VGPR v (pair) holds K = (v>=4?16:0)+2*(v&3)+h + 8*(lane>>4)
__device__ __forceinline__ v16bf load_a_frag(const unsigned short* tile, int lda,
                                             int m0, int k0, int lane) {
    int g = lane >> 4;
    int m = (lane & 15) + m0;
    const unsigned short* row = tile + m * lda;
    Frag16 f;
#pragma unroll
    for (int v = 0; v < 8; ++v) {
        int kk = k0 + ((v >= 4) ? 16 : 0) + ((v & 3) << 1) + (g << 3);
        f.u[v] = *(const unsigned int*)(row + kk);
    }
    return f.v;
}

// B fragment: 32x16 (KxN) bf16, staged TRANSPOSED in LDS as Bt[n][k] with stride ldb.
// Lane (g,n): VGPR v holds K = (v>=4?16:0) + 2*(v&3) + h + 8*g, N = n.
__device__ __forceinline__ v16bf load_b_frag(const unsigned short* tileT, int ldb,
                                             int n0, int k0, int lane) {
    int g = lane >> 4;
    int n = (lane & 15) + n0;
    const unsigned short* row = tileT + n * ldb;
    Frag16 f;
#pragma unroll
    for (int v = 0; v < 8; ++v) {
        int kk = k0 + ((v >= 4) ? 16 : 0) + ((v & 3) << 1) + (g << 3);
        f.u[v] = *(const unsigned int*)(row + kk);
    }
    return f.v;
}

// Async-stage a 64x64 bf16 tile (row-major, global row stride 1024 elems)
// into LDS: 512 B128 chunks, 2 per thread.
__device__ __forceinline__ void stage_tile64(unsigned short* dstLds,
                                             const unsigned short* srcG, int t) {
#pragma unroll
    for (int i = 0; i < 2; ++i) {
        int c = t + i * 256;
        int row = c >> 3, seg = (c & 7) * 8;
        async_ld_b128(dstLds + row * 64 + seg, srcG + (size_t)row * 1024 + seg);
    }
}

// ---------------------------------------------------------------------------
// Kernel 1: fp32 -> bf16 conversion (activations and weights, one-time)
// ---------------------------------------------------------------------------
__global__ __launch_bounds__(256) void cvt_bf16_kernel(const float* __restrict__ x,
                                                       unsigned short* __restrict__ y) {
    int i = (blockIdx.x * 256 + threadIdx.x) * 4;
    float4 v = *(const float4*)(x + i);
    *(unsigned int*)(y + i)     = pack_bf2(v.x, v.y);
    *(unsigned int*)(y + i + 2) = pack_bf2(v.z, v.w);
}

// ---------------------------------------------------------------------------
// GEMM core: out[m][n] = sum_k A[m][k] * W[n][k], both bf16 [.][1024].
// M tile 64, N tile 64, K step 64, double-buffered async global->LDS staging.
// 8 waves: wave w -> (mi = w&3, ni in {2*(w>>2), 2*(w>>2)+1}); 4 WMMA per
// wave per barrier pair; next tile's DMAs run under the current WMMAs.
// ---------------------------------------------------------------------------
struct GemmAcc { v8f a0, a1; };

__device__ __forceinline__ GemmAcc gemm_core(
    const unsigned short* __restrict__ A, const unsigned short* __restrict__ W,
    int m0, int n0, int t,
    unsigned short* As0, unsigned short* As1,
    unsigned short* Bs0, unsigned short* Bs1) {

    const int wave = t >> 5, lane = t & 31;
    const int mi = wave & 3, nb = wave >> 2;

    unsigned short* Acur = As0;
    unsigned short* Anxt = As1;
    unsigned short* Bcur = Bs0;
    unsigned short* Bnxt = Bs1;

    const unsigned short* Ag = A + (size_t)m0 * 1024;
    const unsigned short* Wg = W + (size_t)n0 * 1024;

    GemmAcc acc;
    acc.a0 = (v8f){};
    acc.a1 = (v8f){};

    // Prologue: stage first K-tile
    stage_tile64(Acur, Ag, t);
    stage_tile64(Bcur, Wg, t);

    for (int kt = 0; kt < 1024; kt += 64) {
        if (kt + 64 < 1024) {
            stage_tile64(Anxt, Ag + kt + 64, t);   // prefetch next tile
            stage_tile64(Bnxt, Wg + kt + 64, t);
            wait_async_le4();                      // drain only current tile's DMAs
        } else {
            wait_async0();
        }
        __syncthreads();

#pragma unroll
        for (int ks = 0; ks < 64; ks += 32) {
            v16bf a  = load_a_frag(Acur, 64, mi * 16, ks, lane);
            v16bf b0 = load_b_frag(Bcur, 64, (nb * 2 + 0) * 16, ks, lane);
            v16bf b1 = load_b_frag(Bcur, 64, (nb * 2 + 1) * 16, ks, lane);
            acc.a0 = wmma_bf16(a, b0, acc.a0);
            acc.a1 = wmma_bf16(a, b1, acc.a1);
        }
        __syncthreads();

        unsigned short* tmp = Acur; Acur = Anxt; Anxt = tmp;
        tmp = Bcur; Bcur = Bnxt; Bnxt = tmp;
    }
    return acc;
}

// ---------------------------------------------------------------------------
// Kernel 2: QKV projection GEMM (bf16 out); blockIdx.z selects weight/output.
// ---------------------------------------------------------------------------
__global__ __launch_bounds__(256) void gemm_qkv_kernel(
    const unsigned short* __restrict__ X,
    const unsigned short* __restrict__ Wq, const unsigned short* __restrict__ Wk,
    const unsigned short* __restrict__ Wv,
    unsigned short* __restrict__ Q, unsigned short* __restrict__ K,
    unsigned short* __restrict__ V) {

    const unsigned short* W = (blockIdx.z == 0) ? Wq : (blockIdx.z == 1) ? Wk : Wv;
    unsigned short* O = (blockIdx.z == 0) ? Q : (blockIdx.z == 1) ? K : V;

    __shared__ __align__(16) unsigned short As0[64 * 64];
    __shared__ __align__(16) unsigned short As1[64 * 64];
    __shared__ __align__(16) unsigned short Bs0[64 * 64];
    __shared__ __align__(16) unsigned short Bs1[64 * 64];

    const int t = threadIdx.x;
    const int wave = t >> 5, lane = t & 31;
    const int mi = wave & 3, nb = wave >> 2;
    const int m0 = blockIdx.y * 64, n0 = blockIdx.x * 64;

    GemmAcc acc = gemm_core(X, W, m0, n0, t, As0, As1, Bs0, Bs1);

    // Store C (bf16): lane (g,n), VGPR r -> M = r + 8g
    const int g = lane >> 4, n = lane & 15;
#pragma unroll
    for (int r = 0; r < 8; ++r) {
        int row = m0 + mi * 16 + r + g * 8;
        int c0  = n0 + (nb * 2 + 0) * 16 + n;
        int c1  = n0 + (nb * 2 + 1) * 16 + n;
        O[(size_t)row * 1024 + c0] = f2bf(acc.a0[r]);
        O[(size_t)row * 1024 + c1] = f2bf(acc.a1[r]);
    }
}

// ---------------------------------------------------------------------------
// Kernel 3: flash attention, one block per (batch, head, 64-query tile).
// Q/K/V are bf16 [8192][1024]; head slice at col h*64. Causal.
// ---------------------------------------------------------------------------
__global__ __launch_bounds__(256) void attn_kernel(
    const unsigned short* __restrict__ Q,
    const unsigned short* __restrict__ K,
    const unsigned short* __restrict__ V,
    unsigned short* __restrict__ CTX) {

    __shared__ __align__(16) unsigned short Qt[64 * 64];   // [q][dim]   (A-major)
    __shared__ __align__(16) unsigned short Kt[64 * 64];   // [key][dim] (B^T for Q.K^T)
    __shared__ __align__(16) unsigned short Vt[64 * 64];   // [dim][key] (B^T for P.V)
    __shared__ __align__(16) float          Sb[64 * 64];   // scores, f32
    __shared__ __align__(16) unsigned short Pt[64 * 64];   // probs, bf16 (A-major)
    __shared__ float mrow[64], lrow[64], arow[64];
    __shared__ float red[64 * 4];

    const int t = threadIdx.x;
    const int wave = t >> 5, lane = t & 31;
    const int mi = wave & 3, nb = wave >> 2;
    const int g = lane >> 4, n = lane & 15;

    const int qt = blockIdx.x, h = blockIdx.y, b = blockIdx.z;
    const int q0 = qt * 64;
    const size_t base = ((size_t)b * 2048) * 1024 + (size_t)h * 64;

    // Async-load Q tile
    stage_tile64(Qt, Q + base + (size_t)q0 * 1024, t);
    if (t < 64) { mrow[t] = -3e38f; lrow[t] = 0.f; }

    v8f o0 = {};
    v8f o1 = {};
    wait_async0();
    __syncthreads();

    for (int kb = 0; kb <= qt; ++kb) {      // causal: skip fully-masked key blocks
        const int k0 = kb * 64;
        // Async-stage K tile [key][dim]
        stage_tile64(Kt, K + base + (size_t)k0 * 1024, t);
        // Stage V tile transposed: Vt[dim][key] (coalesced global reads)
        {
            const unsigned short* Vg = V + base + (size_t)k0 * 1024;
#pragma unroll
            for (int i = 0; i < 16; ++i) {
                int e = t + i * 256;
                int key = e >> 6, d = e & 63;
                Vt[d * 64 + key] = Vg[(size_t)key * 1024 + d];
            }
        }
        wait_async0();
        __syncthreads();

        // S = (Q . K^T) * scale  with causal mask, written to Sb (f32)
        {
            v16bf a0 = load_a_frag(Qt, 64, mi * 16, 0, lane);
            v16bf a1 = load_a_frag(Qt, 64, mi * 16, 32, lane);
            int n0a = (nb * 2 + 0) * 16, n0b = (nb * 2 + 1) * 16;
            v8f s0 = {};
            v8f s1 = {};
            s0 = wmma_bf16(a0, load_b_frag(Kt, 64, n0a, 0, lane), s0);
            s0 = wmma_bf16(a1, load_b_frag(Kt, 64, n0a, 32, lane), s0);
            s1 = wmma_bf16(a0, load_b_frag(Kt, 64, n0b, 0, lane), s1);
            s1 = wmma_bf16(a1, load_b_frag(Kt, 64, n0b, 32, lane), s1);
#pragma unroll
            for (int r = 0; r < 8; ++r) {
                int rl = mi * 16 + r + g * 8;
                int qg = q0 + rl;
                int c0 = n0a + n, c1 = n0b + n;
                Sb[rl * 64 + c0] = (k0 + c0 <= qg) ? s0[r] * 0.125f : -3e38f;
                Sb[rl * 64 + c1] = (k0 + c1 <= qg) ? s1[r] * 0.125f : -3e38f;
            }
        }
        __syncthreads();

        // Softmax pass 1: per-row max (4 threads/row x 16 keys)
        {
            int row = t >> 2, seg = t & 3;
            float mx = -3e38f;
#pragma unroll
            for (int j = 0; j < 16; ++j) mx = fmaxf(mx, Sb[row * 64 + seg * 16 + j]);
            red[row * 4 + seg] = mx;
        }
        __syncthreads();
        if (t < 64) {
            float rmax = fmaxf(fmaxf(red[t * 4], red[t * 4 + 1]),
                               fmaxf(red[t * 4 + 2], red[t * 4 + 3]));
            float mold = mrow[t];
            float mnew = fmaxf(mold, rmax);
            mrow[t] = mnew;
            arow[t] = __expf(mold - mnew);
        }
        __syncthreads();

        // Softmax pass 2: P = exp(S - m_new) -> Pt (bf16), partial row sums
        {
            int row = t >> 2, seg = t & 3;
            float mn = mrow[row], sum = 0.f;
#pragma unroll
            for (int j = 0; j < 16; ++j) {
                int idx = row * 64 + seg * 16 + j;
                float p = __expf(Sb[idx] - mn);
                Pt[idx] = f2bf(p);
                sum += p;
            }
            red[row * 4 + seg] = sum;
        }
        // Rescale running O accumulators by alpha (per query row)
#pragma unroll
        for (int r = 0; r < 8; ++r) {
            float a = arow[mi * 16 + r + g * 8];
            o0[r] *= a;
            o1[r] *= a;
        }
        __syncthreads();
        if (t < 64)
            lrow[t] = lrow[t] * arow[t] +
                      (red[t * 4] + red[t * 4 + 1] + red[t * 4 + 2] + red[t * 4 + 3]);

        // O += P . V
        {
            v16bf a0 = load_a_frag(Pt, 64, mi * 16, 0, lane);
            v16bf a1 = load_a_frag(Pt, 64, mi * 16, 32, lane);
            int n0a = (nb * 2 + 0) * 16, n0b = (nb * 2 + 1) * 16;
            o0 = wmma_bf16(a0, load_b_frag(Vt, 64, n0a, 0, lane), o0);
            o0 = wmma_bf16(a1, load_b_frag(Vt, 64, n0a, 32, lane), o0);
            o1 = wmma_bf16(a0, load_b_frag(Vt, 64, n0b, 0, lane), o1);
            o1 = wmma_bf16(a1, load_b_frag(Vt, 64, n0b, 32, lane), o1);
        }
        __syncthreads();
    }

    // Finalize: divide by l, store ctx (bf16) at [token][h*64 + dim]
#pragma unroll
    for (int r = 0; r < 8; ++r) {
        int rl = mi * 16 + r + g * 8;
        float inv_l = 1.0f / lrow[rl];
        int c0 = (nb * 2 + 0) * 16 + n;
        int c1 = (nb * 2 + 1) * 16 + n;
        CTX[base + (size_t)(q0 + rl) * 1024 + c0] = f2bf(o0[r] * inv_l);
        CTX[base + (size_t)(q0 + rl) * 1024 + c1] = f2bf(o1[r] * inv_l);
    }
}

// ---------------------------------------------------------------------------
// Kernel 4: output projection  out[m][n] = sum_k ctx[m][k] * Wo[n][k] + bo[n]
// Double-buffered async-staged bf16 operands, f32 output to d_out.
// ---------------------------------------------------------------------------
__global__ __launch_bounds__(256) void gemm_oproj_kernel(
    const unsigned short* __restrict__ CTX,
    const unsigned short* __restrict__ Wo,
    const float* __restrict__ bo,
    float* __restrict__ out) {

    __shared__ __align__(16) unsigned short As0[64 * 64];
    __shared__ __align__(16) unsigned short As1[64 * 64];
    __shared__ __align__(16) unsigned short Bs0[64 * 64];
    __shared__ __align__(16) unsigned short Bs1[64 * 64];

    const int t = threadIdx.x;
    const int wave = t >> 5, lane = t & 31;
    const int mi = wave & 3, nb = wave >> 2;
    const int m0 = blockIdx.y * 64, n0 = blockIdx.x * 64;

    GemmAcc acc = gemm_core(CTX, Wo, m0, n0, t, As0, As1, Bs0, Bs1);

    const int g = lane >> 4, n = lane & 15;
#pragma unroll
    for (int r = 0; r < 8; ++r) {
        int row = m0 + mi * 16 + r + g * 8;
        int c0  = n0 + (nb * 2 + 0) * 16 + n;
        int c1  = n0 + (nb * 2 + 1) * 16 + n;
        out[(size_t)row * 1024 + c0] = acc.a0[r] + bo[c0];
        out[(size_t)row * 1024 + c1] = acc.a1[r] + bo[c1];
    }
}

// ---------------------------------------------------------------------------
// Host launcher
// ---------------------------------------------------------------------------
extern "C" void kernel_launch(void* const* d_in, const int* in_sizes, int n_in,
                              void* d_out, int out_size, void* d_ws, size_t ws_size,
                              hipStream_t stream) {
    const float* inp = (const float*)d_in[0];
    const float* Wq  = (const float*)d_in[1];
    const float* Wk  = (const float*)d_in[2];
    const float* Wv  = (const float*)d_in[3];
    const float* Wo  = (const float*)d_in[4];
    const float* bo  = (const float*)d_in[5];
    float* out = (float*)d_out;

    const size_t NTOK = 4 * 2048;          // 8192 tokens
    const size_t DIM  = 1024;
    const size_t SZ   = NTOK * DIM;        // 8,388,608 elements
    const size_t WSZ  = DIM * DIM;         // 1,048,576 elements

    unsigned short* Xbf = (unsigned short*)d_ws;   // bf16 workspace tensors
    unsigned short* Qbf = Xbf + SZ;
    unsigned short* Kbf = Qbf + SZ;
    unsigned short* Vbf = Kbf + SZ;
    unsigned short* Ctx = Vbf + SZ;
    unsigned short* Wqb = Ctx + SZ;
    unsigned short* Wkb = Wqb + WSZ;
    unsigned short* Wvb = Wkb + WSZ;
    unsigned short* Wob = Wvb + WSZ;               // total 88 MB

    // 1) one-time conversions to bf16 (activations + all weights)
    cvt_bf16_kernel<<<dim3((unsigned)(SZ / 1024)),  dim3(256), 0, stream>>>(inp, Xbf);
    cvt_bf16_kernel<<<dim3((unsigned)(WSZ / 1024)), dim3(256), 0, stream>>>(Wq, Wqb);
    cvt_bf16_kernel<<<dim3((unsigned)(WSZ / 1024)), dim3(256), 0, stream>>>(Wk, Wkb);
    cvt_bf16_kernel<<<dim3((unsigned)(WSZ / 1024)), dim3(256), 0, stream>>>(Wv, Wvb);
    cvt_bf16_kernel<<<dim3((unsigned)(WSZ / 1024)), dim3(256), 0, stream>>>(Wo, Wob);

    // 2) QKV projections (z selects weight)
    gemm_qkv_kernel<<<dim3(16, 128, 3), dim3(256), 0, stream>>>(
        Xbf, Wqb, Wkb, Wvb, Qbf, Kbf, Vbf);

    // 3) causal flash attention: (32 q-tiles, 16 heads, 4 batches)
    attn_kernel<<<dim3(32, 16, 4), dim3(256), 0, stream>>>(Qbf, Kbf, Vbf, Ctx);

    // 4) output projection + bias -> f32
    gemm_oproj_kernel<<<dim3(16, 128), dim3(256), 0, stream>>>(Ctx, Wob, bo, out);
}
